// PermutedLoss_16071767621852
// MI455X (gfx1250) — matrix operand order repaired and checked by reading, not compiled, not run on hardware
//
#include <hip/hip_runtime.h>

// PIT loss: mean_b min_p sum_i (input[b,perm_p(i)] - target[b,i])^2
// min_p loss = ||x||^2 + ||t||^2 - 2*max_p s_p, with s = P(120x25) * c(25),
// c[i*5+j] = t[i]*x[j]. The 120 scores for 16 rows are one 128x32x16 GEMM:
// 8 x V_WMMA_F32_16X16X32_F16, software-pipelined with ping-pong accumulators
// so the max-folds fill the WMMA->VALU hazard slots (no v_nop chains).

typedef __attribute__((ext_vector_type(16))) _Float16 v16h;
typedef __attribute__((ext_vector_type(8)))  float    v8f;

#define NCOMP 5
#define NPERM 120
#define WAVES_PER_BLOCK 8
#define BLOCK_THREADS (WAVES_PER_BLOCK * 32)
#define MAX_BLOCKS 1024

// Lexicographic permutation #idx of {0..4} via Lehmer code (matches itertools).
__device__ __forceinline__ void perm_from_index(int idx, int p[NCOMP]) {
  int elems[NCOMP] = {0, 1, 2, 3, 4};
  const int fact[NCOMP] = {24, 6, 2, 1, 1};
#pragma unroll
  for (int i = 0; i < NCOMP; ++i) {
    int d = idx / fact[i];
    idx -= d * fact[i];
    p[i] = elems[d];
#pragma unroll
    for (int j = 0; j < NCOMP - 1; ++j)
      if (j >= d) elems[j] = elems[j + 1];
  }
}

// Max over the 8 f32 accumulator elements of one 16x16 tile (per lane).
__device__ __forceinline__ float fold8(const v8f& a) {
  float m01 = fmaxf(a[0], a[1]);
  float m23 = fmaxf(a[2], a[3]);
  float m45 = fmaxf(a[4], a[5]);
  float m67 = fmaxf(a[6], a[7]);
  return fmaxf(fmaxf(m01, m23), fmaxf(m45, m67));
}

__device__ __forceinline__ v8f wmma_pc(const v16h& a, const v16h& b) {
  v8f c;
#pragma unroll
  for (int v = 0; v < 8; ++v) c[v] = 0.0f;  // folds to inline SRC2=0
  return __builtin_amdgcn_wmma_f32_16x16x32_f16(
      /*neg_a=*/false, a, /*neg_b=*/false, b,
      /*c_mod=*/(short)0, c, /*reuse_a=*/false, /*reuse_b=*/false);
}

__global__ __launch_bounds__(BLOCK_THREADS)
void pit_loss_kernel(const float* __restrict__ inp, const float* __restrict__ tgt,
                     float* __restrict__ blockSums, int B) {
  __shared__ float shx[WAVES_PER_BLOCK][16][NCOMP];  // input rows, per-wave slice
  __shared__ float sht[WAVES_PER_BLOCK][16][NCOMP];  // target rows, per-wave slice
  __shared__ float shpart[WAVES_PER_BLOCK];

  const int tid  = threadIdx.x;
  const int w    = tid >> 5;
  const int lane = tid & 31;
  const int half = lane >> 4;   // 0: lanes 0-15, 1: lanes 16-31
  const int col  = lane & 15;   // batch-row index within the 16-row tile

  // ---- Loop-invariant A fragments: 8 tiles of the 128x32 0/1 perm matrix.
  // 16-bit A 16x32 layout (ISA 7.12.2): lane holds row M=lane&15; halves 0..7
  // cover K = 8*half + h, halves 8..15 cover K = 16 + 8*half + (h-8).
  v16h afrag[8];
#pragma unroll
  for (int p = 0; p < 8; ++p) {
    const int mg = p * 16 + col;                 // global perm index 0..127
    int perm[NCOMP];
    perm_from_index(mg < NPERM ? mg : 0, perm);
#pragma unroll
    for (int h = 0; h < 16; ++h) {
      const int k = (h < 8) ? (8 * half + h) : (16 + 8 * half + (h - 8));
      _Float16 v = (_Float16)0.0f;
      if (mg < NPERM && k < NCOMP * NCOMP) {
        const int i = k / NCOMP, j = k - i * NCOMP;
        if (perm[i] == j) v = (_Float16)1.0f;    // P[mg, i*5+j] = [perm(i)==j]
      }
      afrag[p][h] = v;
    }
  }

  const int wavesTotal = gridDim.x * WAVES_PER_BLOCK;
  const int waveId     = blockIdx.x * WAVES_PER_BLOCK + w;
  const int numTiles   = (B + 15) >> 4;

  float waveSum = 0.0f;

  for (int tile = waveId; tile < numTiles; tile += wavesTotal) {
    const int r0 = tile << 4;

    // ---- Stage 16 rows to this wave's LDS slice.
    // lanes 0-15 fetch input rows, lanes 16-31 fetch target rows.
    {
      const int r = r0 + col;
      const float* src = half ? (tgt + (size_t)r * NCOMP)
                              : (inp + (size_t)r * NCOMP);
      float* dst = half ? &sht[w][col][0] : &shx[w][col][0];
      if (r < B) {
#pragma unroll
        for (int j = 0; j < NCOMP; ++j) dst[j] = src[j];
      } else {
#pragma unroll
        for (int j = 0; j < NCOMP; ++j) dst[j] = 0.0f;
      }
    }
    // Same-wave LDS RAW: DS ops are in-order within a wave; the compiler
    // inserts the required s_wait_dscnt. No cross-wave sharing -> no barrier.

    // ---- B fragment: 32x16 f16, element (k, n) = t[n][k/5] * x[n][k%5].
    // element h holds K = 8*half + h (h<8), 16 + 8*half + (h-8) (h>=8), column n = lane&15.
    v16h bfrag;
    const int kb = half * 16;
#pragma unroll
    for (int h = 0; h < 16; ++h) {
      const int k = (h < 8) ? (kb / 2 + h) : (16 + kb / 2 + (h - 8));
      float v = 0.0f;
      if (k < NCOMP * NCOMP) {
        const int i = k / NCOMP, j = k - i * NCOMP;
        v = sht[w][col][i] * shx[w][col][j];
      }
      bfrag[h] = (_Float16)v;
    }

    // ---- 8 WMMAs, ping-pong accumulators: fold tile p while WMMA p+1 runs,
    // filling the 4-slot WMMA->VALU co-exec hazard with useful v_max3 work.
    float smax = -3.0e38f;
    v8f accA, accB;
    accA = wmma_pc(afrag[0], bfrag);
    accB = wmma_pc(afrag[1], bfrag);
    smax = fmaxf(smax, fold8(accA));
    accA = wmma_pc(afrag[2], bfrag);
    smax = fmaxf(smax, fold8(accB));
    accB = wmma_pc(afrag[3], bfrag);
    smax = fmaxf(smax, fold8(accA));
    accA = wmma_pc(afrag[4], bfrag);
    smax = fmaxf(smax, fold8(accB));
    accB = wmma_pc(afrag[5], bfrag);
    smax = fmaxf(smax, fold8(accA));
    accA = wmma_pc(afrag[6], bfrag);
    smax = fmaxf(smax, fold8(accB));
    accB = wmma_pc(afrag[7], bfrag);
    smax = fmaxf(smax, fold8(accA));
    // Tile 7: lanes 16-31 hold M=8..15 -> pad perms 120..127; skip their fold
    // via a select (EXEC stays all-ones around the WMMAs, as required).
    {
      float m7 = fmaxf(smax, fold8(accB));
      smax = half ? smax : m7;
    }
    // Combine the per-column max across the two half-rows of the wave.
    smax = fmaxf(smax, __shfl_xor(smax, 16, 32));

    // ---- Per-row loss: ||x||^2 + ||t||^2 - 2 * max_p s_p.
    float ssq = 0.0f;
#pragma unroll
    for (int j = 0; j < NCOMP; ++j) {
      const float xv = shx[w][col][j];
      const float tv = sht[w][col][j];
      ssq += xv * xv + tv * tv;
    }
    float loss = ssq - 2.0f * smax;
    if (r0 + col >= B) loss = 0.0f;  // padded rows contribute nothing

    // Sum the 16 columns (values duplicated across halves, xor over bits 0..3
    // reduces each group of 16 lanes independently).
#pragma unroll
    for (int d = 1; d < 16; d <<= 1) loss += __shfl_xor(loss, d, 32);
    if (lane == 0) waveSum += loss;
  }

  if (lane == 0) shpart[w] = waveSum;
  __syncthreads();
  if (tid == 0) {
    float s = 0.0f;
#pragma unroll
    for (int i = 0; i < WAVES_PER_BLOCK; ++i) s += shpart[i];
    blockSums[blockIdx.x] = s;
  }
}

__global__ __launch_bounds__(256)
void pit_reduce_kernel(const float* __restrict__ blockSums, int n,
                       float* __restrict__ out, float invB) {
  __shared__ float sh[256];
  float s = 0.0f;
  for (int i = threadIdx.x; i < n; i += 256) s += blockSums[i];
  sh[threadIdx.x] = s;
  __syncthreads();
  for (int off = 128; off > 0; off >>= 1) {
    if ((int)threadIdx.x < off) sh[threadIdx.x] += sh[threadIdx.x + off];
    __syncthreads();
  }
  if (threadIdx.x == 0) out[0] = sh[0] * invB;
}

extern "C" void kernel_launch(void* const* d_in, const int* in_sizes, int n_in,
                              void* d_out, int out_size, void* d_ws, size_t ws_size,
                              hipStream_t stream) {
  const float* inp = (const float*)d_in[0];
  const float* tgt = (const float*)d_in[1];
  float* out       = (float*)d_out;
  float* blockSums = (float*)d_ws;

  const int B = in_sizes[0] / NCOMP;
  const int numTiles = (B + 15) / 16;
  int blocks = (numTiles + WAVES_PER_BLOCK - 1) / WAVES_PER_BLOCK;
  if (blocks > MAX_BLOCKS) blocks = MAX_BLOCKS;
  if (blocks < 1) blocks = 1;

  pit_loss_kernel<<<blocks, BLOCK_THREADS, 0, stream>>>(inp, tgt, blockSums, B);
  pit_reduce_kernel<<<1, 256, 0, stream>>>(blockSums, blocks, out, 1.0f / (float)B);
}
